// GNNModel_45578192945255
// MI455X (gfx1250) — matrix operand order repaired and checked by reading, not compiled, not run on hardware
//
#include <hip/hip_runtime.h>
#include <hip/hip_bf16.h>

// ---------------------------------------------------------------------------
// PointNet++-style GNN forward for MI455X (gfx1250, wave32, WMMA).
// Edge message GEMMs use v_wmma_f32_16x16x32_f16; segment_max uses
// global_atomic_max_num_f32; BN stats fused with ELU.
// ---------------------------------------------------------------------------

typedef __attribute__((ext_vector_type(16))) _Float16 v16h;
typedef __attribute__((ext_vector_type(8)))  _Float16 v8h;
typedef __attribute__((ext_vector_type(8)))  float    v8f;

__device__ __forceinline__ void atomicMaxF32(float* addr, float val) {
    (void)__hip_atomic_fetch_max(addr, val, __ATOMIC_RELAXED, __HIP_MEMORY_SCOPE_AGENT);
}

__device__ __forceinline__ float eluf(float x) {
    return x > 0.0f ? x : (__expf(x) - 1.0f);
}

// ---------------------------------------------------------------------------
// Utility: fill float buffer with a constant.
// ---------------------------------------------------------------------------
__global__ void fill_kernel(float* __restrict__ p, float v, unsigned long long n) {
    unsigned long long i = (unsigned long long)blockIdx.x * blockDim.x + threadIdx.x;
    if (i < n) p[i] = v;
}

// ---------------------------------------------------------------------------
// Weight prep: W [K, Cout] f32 -> Wt [Cop, Kp] f16 (transposed + zero-padded)
// so each WMMA lane reads its B-column as contiguous halves. Bias padded f32.
// ---------------------------------------------------------------------------
__global__ void prep_weights_kernel(const float* __restrict__ W, const float* __restrict__ b,
                                    _Float16* __restrict__ Wt, float* __restrict__ bp,
                                    int K, int Cout, int Kp, int Cop) {
    int idx = blockIdx.x * blockDim.x + threadIdx.x;
    int total = Kp * Cop;
    if (idx < total) {
        int n = idx / Kp;
        int k = idx - n * Kp;
        float v = (k < K && n < Cout) ? W[(size_t)k * Cout + n] : 0.0f;
        Wt[(size_t)n * Kp + k] = (_Float16)v;
    }
    if (idx < Cop) bp[idx] = (idx < Cout) ? b[idx] : 0.0f;
}

// ---------------------------------------------------------------------------
// Edge message GEMM + segment-max scatter.
//   Per workgroup: one tile of 16 edges. A = [16, KP] f16 staged in LDS
//   ([x_src, pos_src - pos_dst], zero padded). COP/16 waves each compute one
//   16x16 output tile with v_wmma_f32_16x16x32_f16, add bias, then
//   atomic-max scatter rows into h[dst, n] (h pre-filled with -1e30).
//
// Fragment layouts used here (contiguous k order, same for A and B):
//   A (16-bit 16x32): lane<16 -> M=lane, halves = K[k0+0..15];
//                     lane>=16 -> K[k0+16..31].
//   B (16-bit 32x16): lanes 0-15 (N=lane) hold K[k0+0..15]; lanes 16-31 hold
//                     K[k0+16..31]  (contiguous in the transposed weight row).
//   C/D f32 16x16:    VGPR v holds M = 8*(lane>=16) + v, N = lane&15.
// ---------------------------------------------------------------------------
template<int CIN, int KP, int COP>
__global__ __launch_bounds__(32 * (COP / 16))
void edge_block_kernel(const float* __restrict__ xin, int xstride,
                       const float* __restrict__ pos,
                       const int* __restrict__ ei, int E,
                       const _Float16* __restrict__ Wt,   // [COP, KP] f16
                       const float* __restrict__ biasP,   // [COP]
                       float* __restrict__ h, int Cout)   // [N, COP], pre-init -1e30
{
    constexpr int CT = COP / 16;
    __shared__ __align__(32) _Float16 As[16 * KP];
    __shared__ int srcS[16];
    __shared__ int dstS[16];

    const int tid = threadIdx.x;
    const int ebase = blockIdx.x * 16;

    if (tid < 16) {
        int e = ebase + tid;
        int s = -1, d = -1;
        if (e < E) { s = ei[e]; d = ei[E + e]; }
        srcS[tid] = s;
        dstS[tid] = d;
    }
    __syncthreads();

    // Cooperative A staging: [x_src (CIN) | pos_src - pos_dst (3) | 0 pad]
    for (int idx = tid; idx < 16 * KP; idx += 32 * CT) {
        int m = idx / KP;
        int k = idx - m * KP;
        int s = srcS[m];
        float v = 0.0f;
        if (s >= 0) {
            if (k < CIN) {
                v = xin[(size_t)s * xstride + k];
            } else if (k < CIN + 3) {
                int d = dstS[m];
                int j = k - CIN;
                v = pos[(size_t)s * 3 + j] - pos[(size_t)d * 3 + j];
            }
        }
        As[idx] = (_Float16)v;
    }
    __syncthreads();

    const int wave = tid >> 5;     // which 16-column tile
    const int lane = tid & 31;
    const int half = lane >> 4;
    const int mn   = lane & 15;
    const int n0   = wave * 16;

    const _Float16* arow = As + mn * KP + half * 16;
    const _Float16* wcol = Wt + (size_t)(n0 + mn) * KP + half * 16;

    v8f acc = {};
#pragma unroll
    for (int k0 = 0; k0 < KP; k0 += 32) {
        v8h alo = *(const v8h*)(arow + k0);
        v8h ahi = *(const v8h*)(arow + k0 + 8);
        v16h a = __builtin_shufflevector(alo, ahi,
                 0, 1, 2, 3, 4, 5, 6, 7, 8, 9, 10, 11, 12, 13, 14, 15);
        v16h b = *(const v16h*)(wcol + k0);
        acc = __builtin_amdgcn_wmma_f32_16x16x32_f16(
                  false, a, false, b, (short)0, acc, false, false);
    }

    const int n = n0 + mn;
    if (n < Cout) {
        const float bias = biasP[n];
#pragma unroll
        for (int v = 0; v < 8; ++v) {
            int m = half * 8 + v;
            int d = dstS[m];
            if (d >= 0) {
                atomicMaxF32(&h[(size_t)d * COP + n], acc[v] + bias);
            }
        }
    }
}

// ---------------------------------------------------------------------------
// Fused ELU (in place) + per-channel sum / sum-of-squares reduction.
// ---------------------------------------------------------------------------
__global__ void elu_stats_kernel(float* __restrict__ h, int N, int C, int Cop,
                                 float* __restrict__ stats) {
    __shared__ float ssum[128];
    __shared__ float ssq[128];
    int t = threadIdx.x;
    if (t < C) { ssum[t] = 0.0f; ssq[t] = 0.0f; }
    __syncthreads();

    int rowsPerBlock = blockDim.x / C;   // C in {8,16,32,64,128}, blockDim 256
    int r = t / C;
    int c = t - r * C;
    float lsum = 0.0f, lsq = 0.0f;
    for (int node = blockIdx.x * rowsPerBlock + r; node < N;
         node += gridDim.x * rowsPerBlock) {
        float v = eluf(h[(size_t)node * Cop + c]);
        h[(size_t)node * Cop + c] = v;
        lsum += v;
        lsq += v * v;
    }
    atomicAdd(&ssum[c], lsum);
    atomicAdd(&ssq[c], lsq);
    __syncthreads();
    if (t < C) {
        atomicAdd(&stats[t], ssum[t]);
        atomicAdd(&stats[C + t], ssq[t]);
    }
}

// ---------------------------------------------------------------------------
// BatchNorm apply (+ optional residual add), compacting stride Cop -> C.
// ---------------------------------------------------------------------------
__global__ void bn_apply_kernel(const float* __restrict__ h, int N, int C, int Cop,
                                const float* __restrict__ stats,
                                const float* __restrict__ gamma,
                                const float* __restrict__ beta,
                                const float* __restrict__ residual,
                                float* __restrict__ out) {
    int idx = blockIdx.x * blockDim.x + threadIdx.x;
    if (idx >= N * C) return;
    int node = idx / C;
    int c = idx - node * C;
    float invN = 1.0f / (float)N;
    float mu = stats[c] * invN;
    float var = stats[C + c] * invN - mu * mu;
    float rstd = rsqrtf(var + 1e-5f);
    float v = gamma[c] * (h[(size_t)node * Cop + c] - mu) * rstd + beta[c];
    if (residual) v += residual[idx];
    out[idx] = v;
}

// ---------------------------------------------------------------------------
// Voxel pool: feature segment-max + position segment-sum + counts.
// ---------------------------------------------------------------------------
__global__ void pool_scatter_kernel(const float* __restrict__ x, int C,
                                    const float* __restrict__ pos,
                                    const int* __restrict__ cluster, int Nin,
                                    float* __restrict__ xp,
                                    float* __restrict__ posn,
                                    float* __restrict__ cnt) {
    int i = blockIdx.x * blockDim.x + threadIdx.x;
    if (i >= Nin) return;
    int c = cluster[i];
    for (int k = 0; k < C; ++k)
        atomicMaxF32(&xp[(size_t)c * C + k], x[(size_t)i * C + k]);
    for (int j = 0; j < 3; ++j)
        atomicAdd(&posn[(size_t)c * 3 + j], pos[(size_t)i * 3 + j]);
    atomicAdd(&cnt[c], 1.0f);
}

__global__ void pool_finalize_kernel(float* __restrict__ posn,
                                     const float* __restrict__ cnt, int Nn) {
    int i = blockIdx.x * blockDim.x + threadIdx.x;
    if (i >= Nn) return;
    float s = fmaxf(cnt[i], 1.0f);
    posn[i * 3 + 0] /= s;
    posn[i * 3 + 1] /= s;
    posn[i * 3 + 2] /= s;
}

// ---------------------------------------------------------------------------
// Final MaxPoolingX (segment_max to [256, 128]) and FC head [16,2048]@[2048,101].
// ---------------------------------------------------------------------------
__global__ void final_scatter_kernel(const float* __restrict__ x,
                                     const int* __restrict__ co, int Nin,
                                     float* __restrict__ xo) {
    int i = blockIdx.x * blockDim.x + threadIdx.x;
    if (i >= Nin) return;
    int c = co[i];
    for (int k = 0; k < 128; ++k)
        atomicMaxF32(&xo[(size_t)c * 128 + k], x[(size_t)i * 128 + k]);
}

__global__ void fc_kernel(const float* __restrict__ xo,
                          const float* __restrict__ w,
                          float* __restrict__ out) {
    int idx = blockIdx.x * blockDim.x + threadIdx.x;
    if (idx >= 16 * 101) return;
    int b = idx / 101;
    int o = idx - b * 101;
    const float* xr = xo + (size_t)b * 2048;
    float s = 0.0f;
    for (int k = 0; k < 2048; ++k) s += xr[k] * w[(size_t)k * 101 + o];
    out[idx] = s;
}

// ---------------------------------------------------------------------------
// Host orchestration.
// ---------------------------------------------------------------------------
extern "C" void kernel_launch(void* const* d_in, const int* in_sizes, int n_in,
                              void* d_out, int out_size, void* d_ws, size_t ws_size,
                              hipStream_t stream) {
    (void)n_in; (void)out_size; (void)ws_size;

    const int N0 = 100000, N1 = 25000, N2 = 6250, N3 = 1600;

    const float* x_in = (const float*)d_in[0];   // [N0, 1]
    const float* pos0 = (const float*)d_in[1];   // [N0, 3]
    const int* ei[4] = {(const int*)d_in[2], (const int*)d_in[3],
                        (const int*)d_in[4], (const int*)d_in[5]};
    const int E[4] = {in_sizes[2] / 2, in_sizes[3] / 2,
                      in_sizes[4] / 2, in_sizes[5] / 2};
    const int* cl[3] = {(const int*)d_in[6], (const int*)d_in[7], (const int*)d_in[8]};
    const int* c_out = (const int*)d_in[9];
    const float* fcw = (const float*)d_in[54];   // params flattened: W,b,gamma,beta per block

    char* ws = (char*)d_ws;
    size_t off = 0;
    auto alloc = [&](size_t bytes) -> void* {
        void* p = ws + off;
        off = (off + bytes + 255) & ~(size_t)255;
        return p;
    };

    struct Cfg { int cin, cout, kp, cop; };
    const Cfg cfg[11] = {
        {1, 8, 32, 16},   {8, 16, 32, 16},  {16, 32, 32, 32},
        {32, 32, 64, 32}, {32, 32, 64, 32}, {32, 64, 64, 64},
        {64, 64, 96, 64}, {64, 64, 96, 64}, {64, 128, 96, 128},
        {128, 128, 160, 128}, {128, 128, 160, 128}
    };

    _Float16* Wt[11];
    float* bp[11];
    for (int i = 0; i < 11; ++i) {
        Wt[i] = (_Float16*)alloc((size_t)cfg[i].kp * cfg[i].cop * sizeof(_Float16));
        bp[i] = (float*)alloc((size_t)cfg[i].cop * sizeof(float));
        int total = cfg[i].kp * cfg[i].cop;
        prep_weights_kernel<<<(total + 255) / 256, 256, 0, stream>>>(
            (const float*)d_in[10 + 4 * i], (const float*)d_in[11 + 4 * i],
            Wt[i], bp[i], cfg[i].cin + 3, cfg[i].cout, cfg[i].kp, cfg[i].cop);
    }

    float* stats = (float*)alloc(256 * sizeof(float));
    float* hbuf  = (float*)alloc((size_t)N0 * 16 * sizeof(float)); // max Nn*Cop

    float* xb0  = (float*)alloc((size_t)N0 * 8 * sizeof(float));
    float* xb1  = (float*)alloc((size_t)N0 * 16 * sizeof(float));
    float* xp1  = (float*)alloc((size_t)N1 * 16 * sizeof(float));
    float* pos1 = (float*)alloc((size_t)N1 * 3 * sizeof(float));
    float* cnt1 = (float*)alloc((size_t)N1 * sizeof(float));
    float* xb2  = (float*)alloc((size_t)N1 * 32 * sizeof(float));
    float* xb3  = (float*)alloc((size_t)N1 * 32 * sizeof(float));
    float* xb4  = (float*)alloc((size_t)N1 * 32 * sizeof(float));
    float* xp2  = (float*)alloc((size_t)N2 * 32 * sizeof(float));
    float* pos2 = (float*)alloc((size_t)N2 * 3 * sizeof(float));
    float* cnt2 = (float*)alloc((size_t)N2 * sizeof(float));
    float* xb5  = (float*)alloc((size_t)N2 * 64 * sizeof(float));
    float* xb6  = (float*)alloc((size_t)N2 * 64 * sizeof(float));
    float* xb7  = (float*)alloc((size_t)N2 * 64 * sizeof(float));
    float* xp3  = (float*)alloc((size_t)N3 * 64 * sizeof(float));
    float* pos3 = (float*)alloc((size_t)N3 * 3 * sizeof(float));
    float* cnt3 = (float*)alloc((size_t)N3 * sizeof(float));
    float* xb8  = (float*)alloc((size_t)N3 * 128 * sizeof(float));
    float* xb9  = (float*)alloc((size_t)N3 * 128 * sizeof(float));
    float* xb10 = (float*)alloc((size_t)N3 * 128 * sizeof(float));
    float* xo   = (float*)alloc((size_t)256 * 128 * sizeof(float));

    auto fill = [&](float* p, float v, size_t n) {
        fill_kernel<<<(unsigned)((n + 255) / 256), 256, 0, stream>>>(p, v, (unsigned long long)n);
    };

    auto run_block = [&](int i, const float* xin, int xs, const float* pos,
                         const int* e, int Ec, int Nn, const float* residual,
                         float* out) {
        const Cfg& c = cfg[i];
        fill(hbuf, -1e30f, (size_t)Nn * c.cop);
        dim3 grid((Ec + 15) / 16), block(32 * (c.cop / 16));
        switch (i) {
            case 0: edge_block_kernel<1, 32, 16><<<grid, block, 0, stream>>>(xin, xs, pos, e, Ec, Wt[i], bp[i], hbuf, c.cout); break;
            case 1: edge_block_kernel<8, 32, 16><<<grid, block, 0, stream>>>(xin, xs, pos, e, Ec, Wt[i], bp[i], hbuf, c.cout); break;
            case 2: edge_block_kernel<16, 32, 32><<<grid, block, 0, stream>>>(xin, xs, pos, e, Ec, Wt[i], bp[i], hbuf, c.cout); break;
            case 3:
            case 4: edge_block_kernel<32, 64, 32><<<grid, block, 0, stream>>>(xin, xs, pos, e, Ec, Wt[i], bp[i], hbuf, c.cout); break;
            case 5: edge_block_kernel<32, 64, 64><<<grid, block, 0, stream>>>(xin, xs, pos, e, Ec, Wt[i], bp[i], hbuf, c.cout); break;
            case 6:
            case 7: edge_block_kernel<64, 96, 64><<<grid, block, 0, stream>>>(xin, xs, pos, e, Ec, Wt[i], bp[i], hbuf, c.cout); break;
            case 8: edge_block_kernel<64, 96, 128><<<grid, block, 0, stream>>>(xin, xs, pos, e, Ec, Wt[i], bp[i], hbuf, c.cout); break;
            default: edge_block_kernel<128, 160, 128><<<grid, block, 0, stream>>>(xin, xs, pos, e, Ec, Wt[i], bp[i], hbuf, c.cout); break;
        }
        fill(stats, 0.0f, (size_t)2 * c.cout);
        int rowsPerBlock = 256 / c.cout;
        int sgrid = (Nn + rowsPerBlock - 1) / rowsPerBlock;
        if (sgrid > 720) sgrid = 720;
        elu_stats_kernel<<<sgrid, 256, 0, stream>>>(hbuf, Nn, c.cout, c.cop, stats);
        int tot = Nn * c.cout;
        bn_apply_kernel<<<(tot + 255) / 256, 256, 0, stream>>>(
            hbuf, Nn, c.cout, c.cop, stats,
            (const float*)d_in[12 + 4 * i], (const float*)d_in[13 + 4 * i],
            residual, out);
    };

    // ---- Level 0 ----
    run_block(0, x_in, 1, pos0, ei[0], E[0], N0, nullptr, xb0);
    run_block(1, xb0, 8, pos0, ei[0], E[0], N0, nullptr, xb1);

    // ---- Pool 0 -> 1 ----
    fill(xp1, -1e30f, (size_t)N1 * 16);
    fill(pos1, 0.0f, (size_t)N1 * 3);
    fill(cnt1, 0.0f, (size_t)N1);
    pool_scatter_kernel<<<(N0 + 255) / 256, 256, 0, stream>>>(xb1, 16, pos0, cl[0], N0, xp1, pos1, cnt1);
    pool_finalize_kernel<<<(N1 + 255) / 256, 256, 0, stream>>>(pos1, cnt1, N1);

    // ---- Level 1 ----
    run_block(2, xp1, 16, pos1, ei[1], E[1], N1, nullptr, xb2);
    run_block(3, xb2, 32, pos1, ei[1], E[1], N1, nullptr, xb3);
    run_block(4, xb3, 32, pos1, ei[1], E[1], N1, xb2, xb4);

    // ---- Pool 1 -> 2 ----
    fill(xp2, -1e30f, (size_t)N2 * 32);
    fill(pos2, 0.0f, (size_t)N2 * 3);
    fill(cnt2, 0.0f, (size_t)N2);
    pool_scatter_kernel<<<(N1 + 255) / 256, 256, 0, stream>>>(xb4, 32, pos1, cl[1], N1, xp2, pos2, cnt2);
    pool_finalize_kernel<<<(N2 + 255) / 256, 256, 0, stream>>>(pos2, cnt2, N2);

    // ---- Level 2 ----
    run_block(5, xp2, 32, pos2, ei[2], E[2], N2, nullptr, xb5);
    run_block(6, xb5, 64, pos2, ei[2], E[2], N2, nullptr, xb6);
    run_block(7, xb6, 64, pos2, ei[2], E[2], N2, xb5, xb7);

    // ---- Pool 2 -> 3 ----
    fill(xp3, -1e30f, (size_t)N3 * 64);
    fill(pos3, 0.0f, (size_t)N3 * 3);
    fill(cnt3, 0.0f, (size_t)N3);
    pool_scatter_kernel<<<(N2 + 255) / 256, 256, 0, stream>>>(xb7, 64, pos2, cl[2], N2, xp3, pos3, cnt3);
    pool_finalize_kernel<<<(N3 + 255) / 256, 256, 0, stream>>>(pos3, cnt3, N3);

    // ---- Level 3 ----
    run_block(8, xp3, 64, pos3, ei[3], E[3], N3, nullptr, xb8);
    run_block(9, xb8, 128, pos3, ei[3], E[3], N3, nullptr, xb9);
    run_block(10, xb9, 128, pos3, ei[3], E[3], N3, xb8, xb10);

    // ---- Head ----
    fill(xo, -1e30f, (size_t)256 * 128);
    final_scatter_kernel<<<(N3 + 255) / 256, 256, 0, stream>>>(xb10, c_out, N3, xo);
    fc_kernel<<<(16 * 101 + 255) / 256, 256, 0, stream>>>(xo, fcw, (float*)d_out);
}
